// RelGAE_29360396436027
// MI455X (gfx1250) — matrix operand order repaired and checked by reading, not compiled, not run on hardware
//
#include <hip/hip_runtime.h>
#include <hip/hip_bf16.h>

typedef __attribute__((ext_vector_type(16))) _Float16 v16h;
typedef __attribute__((ext_vector_type(8)))  _Float16 v8h;
typedef __attribute__((ext_vector_type(4)))  _Float16 v4h;
typedef __attribute__((ext_vector_type(8)))  float    v8f;

#define WAVES 8
#define K1PMAX 128

// ---------------------------------------------------------------------------
// WMMA helpers
// ---------------------------------------------------------------------------
__device__ __forceinline__ v8f wmma_f32_16x16x32_f16(v16h a, v16h b, v8f c) {
  // 8 args: (neg_a, A, neg_b, B, c_mod, C, reuse_a, reuse_b)
  return __builtin_amdgcn_wmma_f32_16x16x32_f16(false, a, false, b, (short)0, c,
                                                false, false);
}

// A-matrix fragment (16x32 f16) from a row-major LDS tile.
// lanes 0-15 : row M=lane,    halves = K[0..7]  and K[16..23] of the 32-chunk
// lanes 16-31: row M=lane-16, halves = K[8..15] and K[24..31]
__device__ __forceinline__ v16h load_a_frag(const _Float16* rowPtr, int kc, int lane) {
  const int base = kc * 32 + ((lane & 16) ? 8 : 0);
  v16h a;
#pragma unroll
  for (int i = 0; i < 8; ++i) {
    a[i]     = rowPtr[base + i];
    a[i + 8] = rowPtr[base + 16 + i];
  }
  return a;
}

__device__ __forceinline__ void store4h(_Float16* p, float4 v) {
  v4h h;
  h[0] = (_Float16)v.x; h[1] = (_Float16)v.y;
  h[2] = (_Float16)v.z; h[3] = (_Float16)v.w;
  *(v4h*)p = h;
}

// ---------------------------------------------------------------------------
// Weight shredding: raw W[K, Ncols] (f32, row-major) -> per-lane B fragments
// layout: frag[((t*kch + kc)*32 + lane)*16 + i] ;  k = kc*32 + 8*(lane>=16) + (i&7) + 16*(i>=8)
// col = t*16 + (lane & 15). Out-of-range k/col -> 0.
// ---------------------------------------------------------------------------
__global__ void prep_frag_kernel(const float* __restrict__ W, int K, int Ncols,
                                 int kch, int Ntiles, _Float16* __restrict__ out) {
  int idx = blockIdx.x * 256 + threadIdx.x;
  int total = Ntiles * kch * 512;
  if (idx >= total) return;
  int i    = idx & 15;
  int lane = (idx >> 4) & 31;
  int kc   = (idx >> 9) % kch;
  int t    = idx / (kch * 512);
  int col  = t * 16 + (lane & 15);
  int k    = kc * 32 + ((lane & 16) ? 8 : 0) + (i & 7) + ((i & 8) ? 16 : 0);
  float v  = (k < K && col < Ncols) ? W[k * Ncols + col] : 0.f;
  out[idx] = (_Float16)v;
}

// ---------------------------------------------------------------------------
// Utility kernels
// ---------------------------------------------------------------------------
__global__ void zero_f32_kernel(float* __restrict__ p, long n) {
  long i = (long)blockIdx.x * 256 + threadIdx.x;
  for (; i < n; i += (long)gridDim.x * 256) p[i] = 0.f;
}

__global__ void count_kernel(const int* __restrict__ tgt, int E, int* __restrict__ cnt) {
  int e = blockIdx.x * 256 + threadIdx.x;
  if (e < E) atomicAdd(&cnt[tgt[e]], 1);
}

__global__ void inv_count_kernel(const int* __restrict__ cnt, int N, float* __restrict__ inv) {
  int n = blockIdx.x * 256 + threadIdx.x;
  if (n < N) {
    int c = cnt[n];
    inv[n] = 1.0f / (float)(c > 0 ? c : 1);
  }
}

// h_new = agg * inv_counts + (Wres ? hold@Wres + bres : hold)
__global__ void node_update_kernel(const float* __restrict__ agg,
                                   const float* __restrict__ inv,
                                   const float* __restrict__ hold, int Fin,
                                   const float* __restrict__ Wres,
                                   const float* __restrict__ bres,
                                   int N, int Fout, float* __restrict__ out) {
  int idx = blockIdx.x * 256 + threadIdx.x;
  if (idx >= N * Fout) return;
  int n = idx / Fout, f = idx - n * Fout;
  float r;
  if (Wres) {
    r = bres[f];
    for (int k = 0; k < Fin; ++k) r += hold[n * Fin + k] * Wres[k * Fout + f];
  } else {
    r = hold[n * Fout + f];
  }
  out[idx] = agg[idx] * inv[n] + r;
}

// ---------------------------------------------------------------------------
// Fused per-edge message block:
//   mode 0 (encoder): msg = [x_i , edge_attr]          (K1 = Fin+3)
//   mode 1 (decoder): msg = [x_i , x_j - x_i]          (K1 = 2*Fin)
//   h = relu(msg @ W1 + b1)   [hidden = 64]
//   m = (h @ W2 + b2)  (+relu unless final)
//   atomic scatter-add of m into agg[tgt]
// Each wave processes 16-edge tiles independently (no cross-wave sync).
// ---------------------------------------------------------------------------
__global__ __launch_bounds__(256)
void edge_block_kernel(const float* __restrict__ h, int Fin,
                       const float* __restrict__ ea,
                       const int* __restrict__ src, const int* __restrict__ tgt, int E,
                       const _Float16* __restrict__ W1f, int K1, int K1p,
                       const float* __restrict__ b1,
                       const _Float16* __restrict__ W2f,
                       const float* __restrict__ b2,
                       int Fout, int mode, int reluOut,
                       float* __restrict__ agg) {
  __shared__ _Float16 sIn[WAVES][16][K1PMAX];
  __shared__ _Float16 sH [WAVES][16][64];
  __shared__ int      sTgt[WAVES][16];

  const int w    = threadIdx.x >> 5;
  const int lane = threadIdx.x & 31;
  const int row  = lane & 15;   // A-row / D-col index
  const int part = lane >> 4;
  const int kch1 = K1p >> 5;
  const int outTiles = (Fout + 15) >> 4;
  const int nTiles   = (E + 15) >> 4;

  for (int tile = blockIdx.x * WAVES + w; tile < nTiles; tile += gridDim.x * WAVES) {
    const int e0 = tile << 4;

    // ---- pad-zero only [K1, K1p) ---------------------------------------
    for (int f = K1 + part; f < K1p; f += 2)
      sIn[w][row][f] = (_Float16)0.f;

    // ---- gather --------------------------------------------------------
    {
      int e = e0 + row; if (e >= E) e = E - 1;
      int t = tgt[e];
      if (part == 0) sTgt[w][row] = t;
      if (Fin == 64) {
        // vectorized: 2 lanes/row, 8x float4 loads each, 4-half LDS packets
        const float4* ht = (const float4*)(h + (size_t)t * 64);
        if (mode == 0) {
#pragma unroll
          for (int i = 0; i < 8; ++i) {
            int q = part + 2 * i;
            store4h(&sIn[w][row][4 * q], ht[q]);
          }
          for (int f = part; f < 3; f += 2)
            sIn[w][row][64 + f] = (_Float16)ea[e * 3 + f];
        } else {
          int s = src[e];
          const float4* hs = (const float4*)(h + (size_t)s * 64);
#pragma unroll
          for (int i = 0; i < 8; ++i) {
            int q = part + 2 * i;
            float4 a = ht[q], b = hs[q];
            store4h(&sIn[w][row][4 * q], a);
            float4 d;
            d.x = b.x - a.x; d.y = b.y - a.y; d.z = b.z - a.z; d.w = b.w - a.w;
            store4h(&sIn[w][row][64 + 4 * q], d);
          }
        }
      } else if (mode == 0) {
        for (int f = part; f < Fin; f += 2)
          sIn[w][row][f] = (_Float16)h[t * Fin + f];
        for (int f = part; f < 3; f += 2)
          sIn[w][row][Fin + f] = (_Float16)ea[e * 3 + f];
      } else {
        int s = src[e];
        for (int f = part; f < Fin; f += 2) {
          float xi = h[t * Fin + f];
          sIn[w][row][f]       = (_Float16)xi;
          sIn[w][row][Fin + f] = (_Float16)(h[s * Fin + f] - xi);
        }
      }
    }
    // DS ops of a wave are in-order on CDNA5; only block compiler reordering.
    asm volatile("" ::: "memory");

    // ---- layer 1: [16 x K1p] @ [K1p x 64] -> ReLU -> sH -----------------
    for (int t = 0; t < 4; ++t) {
      const float bc = b1[t * 16 + row];
      v8f acc;
#pragma unroll
      for (int i = 0; i < 8; ++i) acc[i] = bc;
      for (int kc = 0; kc < kch1; ++kc) {
        v16h a = load_a_frag(&sIn[w][row][0], kc, lane);
        v16h b = ((const v16h*)W1f)[(t * kch1 + kc) * 32 + lane];
        acc = wmma_f32_16x16x32_f16(a, b, acc);
      }
#pragma unroll
      for (int r = 0; r < 8; ++r) {
        float v = acc[r];
        v = v > 0.f ? v : 0.f;
        sH[w][r + part * 8][t * 16 + row] = (_Float16)v;
      }
    }
    asm volatile("" ::: "memory");

    // ---- layer 2: [16 x 64] @ [64 x Fout] -> atomic scatter-add ---------
    for (int t = 0; t < outTiles; ++t) {
      const int col = t * 16 + row;
      const float bc = (col < Fout) ? b2[col] : 0.f;
      v8f acc;
#pragma unroll
      for (int i = 0; i < 8; ++i) acc[i] = bc;
      for (int kc = 0; kc < 2; ++kc) {
        v16h a = load_a_frag(&sH[w][row][0], kc, lane);
        v16h b = ((const v16h*)W2f)[(t * 2 + kc) * 32 + lane];
        acc = wmma_f32_16x16x32_f16(a, b, acc);
      }
      if (col < Fout) {
#pragma unroll
        for (int r = 0; r < 8; ++r) {
          int m = r + part * 8;
          int e = e0 + m;
          if (e < E) {
            float v = acc[r];
            if (reluOut) v = v > 0.f ? v : 0.f;
            atomicAdd(&agg[sTgt[w][m] * Fout + col], v);
          }
        }
      }
    }
  }
}

// ---------------------------------------------------------------------------
// Edge predictor: x_cat = [min(xs,xt), max(xs,xt)] (4)
//   h  = relu(fc1(x_cat));  h2 = relu(fc2(h) + h)
//   out = fc3(h2) + fc_direct(x_cat)   -> [E, 3]
// ---------------------------------------------------------------------------
__global__ __launch_bounds__(256)
void edge_pred_kernel(const float* __restrict__ xe,
                      const int* __restrict__ src, const int* __restrict__ tgt, int E,
                      const _Float16* __restrict__ f1, const float* __restrict__ b1,
                      const _Float16* __restrict__ f2, const float* __restrict__ b2,
                      const _Float16* __restrict__ f3, const float* __restrict__ b3,
                      const _Float16* __restrict__ fd, const float* __restrict__ bd,
                      float* __restrict__ out) {
  __shared__ _Float16 sIn[WAVES][16][32];
  __shared__ _Float16 sH [WAVES][16][64];
  __shared__ _Float16 sH2[WAVES][16][64];

  const int w    = threadIdx.x >> 5;
  const int lane = threadIdx.x & 31;
  const int row  = lane & 15;
  const int part = lane >> 4;
  const int nTiles = (E + 15) >> 4;

  for (int tile = blockIdx.x * WAVES + w; tile < nTiles; tile += gridDim.x * WAVES) {
    const int e0 = tile << 4;
    {
      v8h z = {};
      _Float16* p = &sIn[w][row][part * 16];
      ((v8h*)p)[0] = z;
      ((v8h*)p)[1] = z;
    }
    {
      int e = e0 + row; if (e >= E) e = E - 1;
      int s = src[e], t = tgt[e];
      int f = part;               // f in {0,1}
      float a = xe[s * 2 + f], b = xe[t * 2 + f];
      sIn[w][row][f]     = (_Float16)fminf(a, b);
      sIn[w][row][2 + f] = (_Float16)fmaxf(a, b);
    }
    asm volatile("" ::: "memory");

    // fc1: [16x32pad] @ [32x64] -> relu -> sH
    for (int t = 0; t < 4; ++t) {
      const float bc = b1[t * 16 + row];
      v8f acc;
#pragma unroll
      for (int i = 0; i < 8; ++i) acc[i] = bc;
      v16h a = load_a_frag(&sIn[w][row][0], 0, lane);
      v16h b = ((const v16h*)f1)[t * 32 + lane];
      acc = wmma_f32_16x16x32_f16(a, b, acc);
#pragma unroll
      for (int r = 0; r < 8; ++r) {
        float v = acc[r]; v = v > 0.f ? v : 0.f;
        sH[w][r + part * 8][t * 16 + row] = (_Float16)v;
      }
    }
    asm volatile("" ::: "memory");

    // fc2 + skip: relu(fc2(h) + h) -> sH2
    for (int t = 0; t < 4; ++t) {
      const int col = t * 16 + row;
      const float bc = b2[col];
      v8f acc;
#pragma unroll
      for (int i = 0; i < 8; ++i) acc[i] = bc;
      for (int kc = 0; kc < 2; ++kc) {
        v16h a = load_a_frag(&sH[w][row][0], kc, lane);
        v16h b = ((const v16h*)f2)[(t * 2 + kc) * 32 + lane];
        acc = wmma_f32_16x16x32_f16(a, b, acc);
      }
#pragma unroll
      for (int r = 0; r < 8; ++r) {
        int m = r + part * 8;
        float v = acc[r] + (float)sH[w][m][col];
        v = v > 0.f ? v : 0.f;
        sH2[w][m][col] = (_Float16)v;
      }
    }
    asm volatile("" ::: "memory");

    // fc3(h2) + fc_direct(x_cat), cols 0..2 valid
    {
      const int col = row;
      v8f acc3;
#pragma unroll
      for (int i = 0; i < 8; ++i) acc3[i] = (col < 3) ? b3[col] : 0.f;
      for (int kc = 0; kc < 2; ++kc) {
        v16h a = load_a_frag(&sH2[w][row][0], kc, lane);
        v16h b = ((const v16h*)f3)[kc * 32 + lane];
        acc3 = wmma_f32_16x16x32_f16(a, b, acc3);
      }
      v8f accd;
#pragma unroll
      for (int i = 0; i < 8; ++i) accd[i] = (col < 3) ? bd[col] : 0.f;
      {
        v16h a = load_a_frag(&sIn[w][row][0], 0, lane);
        v16h b = ((const v16h*)fd)[lane];
        accd = wmma_f32_16x16x32_f16(a, b, accd);
      }
      if (col < 3) {
#pragma unroll
        for (int r = 0; r < 8; ++r) {
          int m = r + part * 8;
          int e = e0 + m;
          if (e < E) out[e * 3 + col] = acc3[r] + accd[r];
        }
      }
    }
  }
}

// ---------------------------------------------------------------------------
// Host orchestration
// ---------------------------------------------------------------------------
extern "C" void kernel_launch(void* const* d_in, const int* in_sizes, int n_in,
                              void* d_out, int out_size, void* d_ws, size_t ws_size,
                              hipStream_t stream) {
  const int N = in_sizes[0];          // x is [N, 1]
  const int E = in_sizes[2] / 2;      // edge_index is [2, E]

  const float* x   = (const float*)d_in[0];
  const float* ea  = (const float*)d_in[1];
  const int*   ei  = (const int*)d_in[2];
  const int* src = ei;
  const int* tgt = ei + E;

  // params: jax pytree flatten order (dict keys sorted): dec, edge_pred, enc
  const float* dec0_m1W = (const float*)d_in[3];
  const float* dec0_m1b = (const float*)d_in[4];
  const float* dec0_m2W = (const float*)d_in[5];
  const float* dec0_m2b = (const float*)d_in[6];
  const float* dec0_rW  = (const float*)d_in[7];
  const float* dec0_rb  = (const float*)d_in[8];
  const float* dec1_m1W = (const float*)d_in[9];
  const float* dec1_m1b = (const float*)d_in[10];
  const float* dec1_m2W = (const float*)d_in[11];
  const float* dec1_m2b = (const float*)d_in[12];
  const float* dec1_rW  = (const float*)d_in[13];
  const float* dec1_rb  = (const float*)d_in[14];
  const float* ep_f1W   = (const float*)d_in[15];
  const float* ep_f1b   = (const float*)d_in[16];
  const float* ep_f2W   = (const float*)d_in[17];
  const float* ep_f2b   = (const float*)d_in[18];
  const float* ep_f3W   = (const float*)d_in[19];
  const float* ep_f3b   = (const float*)d_in[20];
  const float* ep_fdW   = (const float*)d_in[21];
  const float* ep_fdb   = (const float*)d_in[22];
  const float* enc0_m1W = (const float*)d_in[23];
  const float* enc0_m1b = (const float*)d_in[24];
  const float* enc0_m2W = (const float*)d_in[25];
  const float* enc0_m2b = (const float*)d_in[26];
  const float* enc0_rW  = (const float*)d_in[27];
  const float* enc0_rb  = (const float*)d_in[28];
  const float* enc1_m1W = (const float*)d_in[29];
  const float* enc1_m1b = (const float*)d_in[30];
  const float* enc1_m2W = (const float*)d_in[31];
  const float* enc1_m2b = (const float*)d_in[32];
  const float* enc2_m1W = (const float*)d_in[33];
  const float* enc2_m1b = (const float*)d_in[34];
  const float* enc2_m2W = (const float*)d_in[35];
  const float* enc2_m2b = (const float*)d_in[36];
  const float* enc2_rW  = (const float*)d_in[37];
  const float* enc2_rb  = (const float*)d_in[38];

  // ---- carve workspace -------------------------------------------------
  char* wp = (char*)d_ws;
  auto carve = [&](size_t bytes) -> void* {
    void* r = (void*)wp;
    wp += (bytes + 255) & ~(size_t)255;
    return r;
  };
  int*   counts = (int*)  carve((size_t)N * 4);
  float* inv    = (float*)carve((size_t)N * 4);
  float* hbuf0  = (float*)carve((size_t)N * 64 * 4);
  float* hbuf1  = (float*)carve((size_t)N * 64 * 4);
  float* agg    = (float*)carve((size_t)N * 64 * 4);
  float* xenc   = (float*)carve((size_t)N * 2 * 4);

  auto carveFrag = [&](int K1p, int Nt) -> _Float16* {
    return (_Float16*)carve((size_t)Nt * (K1p / 32) * 512 * 2);
  };
  _Float16* F_enc0_1 = carveFrag(32, 4);
  _Float16* F_enc0_2 = carveFrag(64, 4);
  _Float16* F_enc1_1 = carveFrag(96, 4);
  _Float16* F_enc1_2 = carveFrag(64, 4);
  _Float16* F_enc2_1 = carveFrag(96, 4);
  _Float16* F_enc2_2 = carveFrag(64, 1);
  _Float16* F_dec0_1 = carveFrag(32, 4);
  _Float16* F_dec0_2 = carveFrag(64, 4);
  _Float16* F_dec1_1 = carveFrag(128, 4);
  _Float16* F_dec1_2 = carveFrag(64, 1);
  _Float16* F_f1     = carveFrag(32, 4);
  _Float16* F_f2     = carveFrag(64, 4);
  _Float16* F_f3     = carveFrag(64, 1);
  _Float16* F_fd     = carveFrag(32, 1);

  // ---- counts / inv_counts --------------------------------------------
  zero_f32_kernel<<<256, 256, 0, stream>>>((float*)counts, N);
  count_kernel<<<(E + 255) / 256, 256, 0, stream>>>(tgt, E, counts);
  inv_count_kernel<<<(N + 255) / 256, 256, 0, stream>>>(counts, N, inv);

  // ---- shred weights into WMMA B-fragment layout -----------------------
  auto prep = [&](const float* W, int K, int Ncols, int K1p, int Nt, _Float16* out) {
    int total = Nt * (K1p / 32) * 512;
    prep_frag_kernel<<<(total + 255) / 256, 256, 0, stream>>>(W, K, Ncols, K1p / 32, Nt, out);
  };
  prep(enc0_m1W,  4, 64,  32, 4, F_enc0_1);
  prep(enc0_m2W, 64, 64,  64, 4, F_enc0_2);
  prep(enc1_m1W, 67, 64,  96, 4, F_enc1_1);
  prep(enc1_m2W, 64, 64,  64, 4, F_enc1_2);
  prep(enc2_m1W, 67, 64,  96, 4, F_enc2_1);
  prep(enc2_m2W, 64,  2,  64, 1, F_enc2_2);
  prep(dec0_m1W,  4, 64,  32, 4, F_dec0_1);
  prep(dec0_m2W, 64, 64,  64, 4, F_dec0_2);
  prep(dec1_m1W,128, 64, 128, 4, F_dec1_1);
  prep(dec1_m2W, 64,  1,  64, 1, F_dec1_2);
  prep(ep_f1W,    4, 64,  32, 4, F_f1);
  prep(ep_f2W,   64, 64,  64, 4, F_f2);
  prep(ep_f3W,   64,  3,  64, 1, F_f3);
  prep(ep_fdW,    4,  3,  32, 1, F_fd);

  const int EB = 1024;  // edge-kernel blocks (8 waves each)

  auto runBlock = [&](const float* hin, int Fin, int mode,
                      const _Float16* W1f, int K1, int K1p, const float* b1,
                      const _Float16* W2f, const float* b2, int Fout, int relu,
                      const float* Wres, const float* bres, float* hout) {
    zero_f32_kernel<<<2048, 256, 0, stream>>>(agg, (long)N * Fout);
    edge_block_kernel<<<EB, 256, 0, stream>>>(hin, Fin, ea, src, tgt, E,
                                              W1f, K1, K1p, b1, W2f, b2,
                                              Fout, mode, relu, agg);
    node_update_kernel<<<(N * Fout + 255) / 256, 256, 0, stream>>>(
        agg, inv, hin, Fin, Wres, bres, N, Fout, hout);
  };

  // encoder
  runBlock(x,     1, 0, F_enc0_1,   4,  32, enc0_m1b, F_enc0_2, enc0_m2b, 64, 1, enc0_rW, enc0_rb, hbuf0);
  runBlock(hbuf0,64, 0, F_enc1_1,  67,  96, enc1_m1b, F_enc1_2, enc1_m2b, 64, 1, nullptr, nullptr, hbuf1);
  runBlock(hbuf1,64, 0, F_enc2_1,  67,  96, enc2_m1b, F_enc2_2, enc2_m2b,  2, 1, enc2_rW, enc2_rb, xenc);
  // decoder (final block: no relu on message) -> x_recon in d_out[0..N)
  runBlock(xenc,  2, 1, F_dec0_1,   4,  32, dec0_m1b, F_dec0_2, dec0_m2b, 64, 1, dec0_rW, dec0_rb, hbuf0);
  runBlock(hbuf0,64, 1, F_dec1_1, 128, 128, dec1_m1b, F_dec1_2, dec1_m2b,  1, 0, dec1_rW, dec1_rb, (float*)d_out);

  // edge predictor -> d_out[N .. N+3E)
  edge_pred_kernel<<<EB, 256, 0, stream>>>(xenc, src, tgt, E,
                                           F_f1, ep_f1b, F_f2, ep_f2b,
                                           F_f3, ep_f3b, F_fd, ep_fdb,
                                           (float*)d_out + N);
}
